// TransformerClassifier_40218073759900
// MI455X (gfx1250) — matrix operand order repaired and checked by reading, not compiled, not run on hardware
//
// TransformerClassifier forward for MI455X (gfx1250, wave32, WMMA).
//
// Roofline: ~0.48 TFLOP vs ~100 MB HBM traffic -> compute bound; must hit the
// matrix pipes. All GEMMs (QKV, O, FF1, FF2) and both attention batched
// matmuls (Q.K^T, P.V) run on v_wmma_f32_16x16x32_f16 (f16 in, f32 acc).
// Weights fit in the 192 MB L2, so global loads + LDS staging suffice.
// Contiguous tile staging uses CDNA5 GLOBAL_LOAD_ASYNC_TO_LDS_B128
// (ASYNCcnt-tracked, no VGPR bounce); transposing stages stay synchronous.
#include <hip/hip_runtime.h>
#include <cmath>

typedef _Float16 h16;
typedef __attribute__((ext_vector_type(16))) _Float16 v16h;
typedef __attribute__((ext_vector_type(8)))  _Float16 v8h;
typedef __attribute__((ext_vector_type(8)))  float    v8f;
typedef __attribute__((ext_vector_type(4)))  int      v4i;

#define TC_B   32
#define TC_L   512
#define TC_D   512
#define TC_H   8
#define TC_NL  4
#define TC_DK  64
#define TC_DFF 2048
#define TC_M   (TC_B*TC_L)   // 16384 tokens

// ---- CDNA5 async global->LDS staging (guarded; falls back to sync copies)
#if defined(__has_builtin)
#if __has_builtin(__builtin_amdgcn_global_load_async_to_lds_b128) && \
    __has_builtin(__builtin_amdgcn_s_wait_asynccnt)
#define TC_ASYNC 1
#endif
#endif
#ifndef TC_ASYNC
#define TC_ASYNC 0
#endif

__device__ __forceinline__ void stage16(const h16* g, h16* l){
#if TC_ASYNC
  // builtin signature (from hipcc diagnostic): (AS1 v4i*, AS3 v4i*, Ii, Ii)
  __builtin_amdgcn_global_load_async_to_lds_b128(
      (__attribute__((address_space(1))) v4i*)
        (__attribute__((address_space(1))) void*)g,
      (__attribute__((address_space(3))) v4i*)
        (__attribute__((address_space(3))) void*)l,
      0, 0);
#else
  *(v8h*)l = *(const v8h*)g;
#endif
}
__device__ __forceinline__ void stage_wait(){
#if TC_ASYNC
  __builtin_amdgcn_s_wait_asynccnt(0);
#endif
}

__device__ __forceinline__ float gelu_exact(float x){
  return 0.5f*x*(1.0f+erff(x*0.70710678118654752f));
}

// Build a v16h WMMA operand from two 16-byte chunks (LDS or global).
__device__ __forceinline__ v16h load_frag(const h16* p0, const h16* p1){
  v8h lo = *(const v8h*)p0;
  v8h hi = *(const v8h*)p1;
  v16h r;
#pragma unroll
  for (int i=0;i<8;++i){ r[i]=lo[i]; r[i+8]=hi[i]; }
  return r;
}

// ---------------------------------------------------------------- embedding
__global__ void __launch_bounds__(128)
embed_kernel(const int* __restrict__ ids, const float* __restrict__ emb,
             const float* __restrict__ pe, float* __restrict__ h)
{
  const int tok = blockIdx.x;
  const int l   = tok & (TC_L-1);
  const int id  = ids[tok];
  const int d   = threadIdx.x*4;
  const float4 e = *(const float4*)(emb + (size_t)id*TC_D + d);
  const float4 p = *(const float4*)(pe  + (size_t)l *TC_D + d);
  const float s = 22.627416997969522f;   // sqrt(512)
  float4 r; r.x=e.x*s+p.x; r.y=e.y*s+p.y; r.z=e.z*s+p.z; r.w=e.w*s+p.w;
  *(float4*)(h + (size_t)tok*TC_D + d) = r;
}

// ---------------------------------------------------------------- f32->f16
__global__ void __launch_bounds__(256)
f32_to_f16_kernel(const float* __restrict__ s, h16* __restrict__ d, int n)
{
  for (int i = blockIdx.x*blockDim.x + threadIdx.x; i < n;
       i += gridDim.x*blockDim.x)
    d[i] = (h16)s[i];
}

// ---------------------------------------------------------------- layernorm
template<typename OUT>
__global__ void __launch_bounds__(256)
layernorm_kernel(const float* __restrict__ x, const float* __restrict__ g,
                 const float* __restrict__ bb, OUT* __restrict__ out)
{
  const int row = blockIdx.x, tid = threadIdx.x;
  const float* xr = x + (size_t)row*TC_D;
  const float x0 = xr[tid], x1 = xr[tid+256];
  float s = x0 + x1;
#pragma unroll
  for (int off=16; off; off>>=1) s += __shfl_xor(s, off, 32);
  __shared__ float sh[8], sh2[8];
  if ((tid&31)==0) sh[tid>>5] = s;
  __syncthreads();
  float tot = 0.f;
#pragma unroll
  for (int i=0;i<8;++i) tot += sh[i];
  const float mu = tot * (1.f/(float)TC_D);
  const float d0 = x0-mu, d1 = x1-mu;
  float vs = d0*d0 + d1*d1;
#pragma unroll
  for (int off=16; off; off>>=1) vs += __shfl_xor(vs, off, 32);
  if ((tid&31)==0) sh2[tid>>5] = vs;
  __syncthreads();
  float vt = 0.f;
#pragma unroll
  for (int i=0;i<8;++i) vt += sh2[i];
  const float rstd = rsqrtf(vt*(1.f/(float)TC_D) + 1e-5f);
  out[(size_t)row*TC_D + tid      ] = (OUT)(d0*rstd*g[tid]     + bb[tid]);
  out[(size_t)row*TC_D + tid + 256] = (OUT)(d1*rstd*g[tid+256] + bb[tid+256]);
}

// ---------------------------------------------------------------- WMMA GEMM
// C[M,N] = epilogue(A[M,K](f16) * W[K,N](f16) + bias). Block tile 128x64,
// 8 waves, each wave: one 16-row strip x 64 cols = 4 accumulators.
enum { EPI_F16=0, EPI_GELU=1, EPI_RES=2 };

template<int EPI>
__global__ void __launch_bounds__(256)
gemm_f16_wmma(const h16* __restrict__ A, const h16* __restrict__ W,
              const float* __restrict__ bias, const float* __restrict__ resid,
              h16* __restrict__ out16, float* __restrict__ out32,
              int M, int N, int K)
{
  __shared__ h16 As[128][40];   // 128 rows x 32 K (pad 8 -> 80B rows)
  __shared__ h16 Bt[64][40];    // transposed: [n][k]
  const int tid  = threadIdx.x;
  const int wave = tid>>5, lane = tid&31;
  const int lg = lane>>4, ln = lane&15;
  const int m0 = blockIdx.x*128, n0 = blockIdx.y*64;

  v8f acc[4] = {};

  for (int k0 = 0; k0 < K; k0 += 32){
    __syncthreads();
    // stage A tile (128x32): 2 iters x 256 lanes x 16B -> async b128 to LDS
#pragma unroll
    for (int it=0; it<2; ++it){
      const int r = (tid>>2) + it*64;
      const int c = (tid&3)*8;
      stage16(A + (size_t)(m0+r)*K + k0 + c, &As[r][c]);
    }
    // stage W tile (32x64) transposed into Bt[n][k] (scatter: sync path)
    {
      const int kk = tid>>3;
      const int nc = (tid&7)*8;
      const v8h w = *(const v8h*)(W + (size_t)(k0+kk)*N + n0 + nc);
#pragma unroll
      for (int i=0;i<8;++i) Bt[nc+i][kk] = w[i];
    }
    if (k0 + 32 < K){  // -> global_prefetch_b8
      __builtin_prefetch(A + (size_t)(m0+(tid>>2))*K + k0 + 32, 0, 3);
      __builtin_prefetch(W + (size_t)(k0+32+(tid>>3))*N + n0,   0, 3);
    }
    stage_wait();
    __syncthreads();

    // A frag: lane ln = row, lg selects K chunks {0..7,16..23} / {8..15,24..31}
    const int arow = wave*16 + ln;
    const v16h a = load_frag(&As[arow][lg*8], &As[arow][16 + lg*8]);
#pragma unroll
    for (int j=0;j<4;++j){
      // B frag: lane ln = column, lg selects K chunks {0..7,16..23} / {8..15,24..31}
      const v16h b = load_frag(&Bt[j*16+ln][lg*8], &Bt[j*16+ln][16 + lg*8]);
      acc[j] = __builtin_amdgcn_wmma_f32_16x16x32_f16(
                 false, a, false, b, (short)0, acc[j], false, false);
    }
  }

  // C/D layout: row m = vgpr + 8*(lane/16), col n = lane%16
  const int rbase = m0 + wave*16 + lg*8;
#pragma unroll
  for (int j=0;j<4;++j){
    const int col = n0 + j*16 + ln;
    const float bv = bias[col];
#pragma unroll
    for (int i=0;i<8;++i){
      const size_t idx = (size_t)(rbase+i)*N + col;
      const float v = acc[j][i] + bv;
      if (EPI == EPI_GELU)     out16[idx] = (h16)gelu_exact(v);
      else if (EPI == EPI_RES) out32[idx] = resid[idx] + v;
      else                     out16[idx] = (h16)v;
    }
  }
}

// ---------------------------------------------------------------- attention
// grid = (B*H, L/128); 8 waves, each owns 16 query rows; flash-style online
// softmax over 32-key blocks. Q,K,V,O are f16 [B, L, H*DK].
__global__ void __launch_bounds__(256)
attention_kernel(const h16* __restrict__ Q, const h16* __restrict__ Kk,
                 const h16* __restrict__ V, const unsigned char* __restrict__ mask,
                 h16* __restrict__ O)
{
  __shared__ h16 Ks[32][64];     // key block, row-major [key][feat]
  __shared__ h16 Vt[64][40];     // value block transposed [feat][key]
  __shared__ h16 Pt[8][16][40];  // per-wave P tile bounce buffer

  const int tid  = threadIdx.x;
  const int wave = tid>>5, lane = tid&31;
  const int lg = lane>>4, ln = lane&15;
  const int b  = blockIdx.x >> 3;
  const int hh = blockIdx.x & 7;
  const int q0 = blockIdx.y*128 + wave*16;
  const size_t headoff = (size_t)b*TC_L*TC_D + (size_t)hh*TC_DK;

  // Q fragments for this wave's 16 rows, both 32-wide K-steps (DK=64)
  v16h aq[2];
#pragma unroll
  for (int s=0;s<2;++s){
    const h16* qp = Q + headoff + (size_t)(q0+ln)*TC_D + s*32;
    aq[s] = load_frag(qp + lg*8, qp + 16 + lg*8);
  }

  const float NEGINF = -__builtin_inff();
  float rmax[8], rsum[8];
  v8f ctx[4] = {};
#pragma unroll
  for (int i=0;i<8;++i){ rmax[i]=NEGINF; rsum[i]=0.f; }

  for (int kb = 0; kb < TC_L; kb += 32){
    __syncthreads();
    { // cooperative stage: 32 keys x 64 feats; K async, V transposed (sync)
      const int key = tid>>3, c = (tid&7)*8;
      stage16(Kk + headoff + (size_t)(kb+key)*TC_D + c, &Ks[key][c]);
      const v8h vw = *(const v8h*)(V + headoff + (size_t)(kb+key)*TC_D + c);
#pragma unroll
      for (int i=0;i<8;++i) Vt[c+i][key] = vw[i];
    }
    stage_wait();
    __syncthreads();

    // S = Q.K^T for two 16-key tiles, K-dim in 2 steps of 32
    v8f sc[2] = {};
#pragma unroll
    for (int t=0;t<2;++t){
#pragma unroll
      for (int s=0;s<2;++s){
        const v16h b = load_frag(&Ks[t*16+ln][s*32 + lg*8],
                                 &Ks[t*16+ln][s*32 + 16 + lg*8]);
        sc[t] = __builtin_amdgcn_wmma_f32_16x16x32_f16(
                  false, aq[s], false, b, (short)0, sc[t], false, false);
      }
    }

    const bool mk0 = mask[(size_t)b*TC_L + kb +      ln] != 0;
    const bool mk1 = mask[(size_t)b*TC_L + kb + 16 + ln] != 0;
#pragma unroll
    for (int i=0;i<8;++i){
      const float s0 = mk0 ? NEGINF : sc[0][i]*0.125f;  // 1/sqrt(DK)
      const float s1 = mk1 ? NEGINF : sc[1][i]*0.125f;
      float x = fmaxf(s0, s1);
#pragma unroll
      for (int off=1; off<16; off<<=1) x = fmaxf(x, __shfl_xor(x, off, 32));
      const float mnew = fmaxf(rmax[i], x);
      float alpha, p0, p1;
      if (mnew == NEGINF){ alpha = 1.f; p0 = 0.f; p1 = 0.f; }
      else {
        alpha = __expf(rmax[i] - mnew);
        p0 = (s0 == NEGINF) ? 0.f : __expf(s0 - mnew);
        p1 = (s1 == NEGINF) ? 0.f : __expf(s1 - mnew);
      }
      rmax[i] = mnew;
      float ps = p0 + p1;
#pragma unroll
      for (int off=1; off<16; off<<=1) ps += __shfl_xor(ps, off, 32);
      rsum[i] = rsum[i]*alpha + ps;
#pragma unroll
      for (int j=0;j<4;++j) ctx[j][i] *= alpha;
      Pt[wave][lg*8 + i][ln]      = (h16)p0;   // C layout -> LDS row-major
      Pt[wave][lg*8 + i][16 + ln] = (h16)p1;
    }

    // re-fragment P as an A operand, ctx += P.V
    const v16h ap = load_frag(&Pt[wave][ln][lg*8], &Pt[wave][ln][16 + lg*8]);
#pragma unroll
    for (int j=0;j<4;++j){
      const v16h bv = load_frag(&Vt[j*16+ln][lg*8], &Vt[j*16+ln][16 + lg*8]);
      ctx[j] = __builtin_amdgcn_wmma_f32_16x16x32_f16(
                 false, ap, false, bv, (short)0, ctx[j], false, false);
    }
  }

#pragma unroll
  for (int j=0;j<4;++j){
#pragma unroll
    for (int i=0;i<8;++i){
      const int mrow = q0 + lg*8 + i;
      O[headoff + (size_t)mrow*TC_D + j*16 + ln] = (h16)(ctx[j][i] / rsum[i]);
    }
  }
}

// ---------------------------------------------------------------- pool
__global__ void __launch_bounds__(256)
pool_kernel(const float* __restrict__ hn, float* __restrict__ pooled)
{
  const int idx = blockIdx.x*256 + threadIdx.x;     // b*D + d
  const int b = idx >> 9, d = idx & 511;
  const float* p = hn + (size_t)b*TC_L*TC_D + d;
  float s = 0.f;
  for (int l=0; l<TC_L; ++l) s += p[(size_t)l*TC_D];
  pooled[idx] = s * (1.f/(float)TC_L);
}

// ---------------------------------------------------------------- classifier
__global__ void __launch_bounds__(256)
classifier_kernel(const float* __restrict__ pooled, const float* __restrict__ W1,
                  const float* __restrict__ b1, const float* __restrict__ W2,
                  const float* __restrict__ b2, float* __restrict__ out)
{
  const int b = blockIdx.x, tid = threadIdx.x;
  __shared__ float sp[TC_D];
  __shared__ float t1[TC_D];
  __shared__ float r0[8], r1[8];
  sp[tid]     = pooled[(size_t)b*TC_D + tid];
  sp[tid+256] = pooled[(size_t)b*TC_D + tid + 256];
  __syncthreads();
  for (int d = tid; d < TC_D; d += 256){
    float acc = b1[d];
    for (int k=0;k<TC_D;++k) acc += sp[k]*W1[(size_t)k*TC_D + d];
    t1[d] = gelu_exact(acc);
  }
  __syncthreads();
  float p0 = 0.f, p1 = 0.f;
  for (int d = tid; d < TC_D; d += 256){
    p0 += t1[d]*W2[d*2+0];
    p1 += t1[d]*W2[d*2+1];
  }
#pragma unroll
  for (int off=16; off; off>>=1){
    p0 += __shfl_xor(p0, off, 32);
    p1 += __shfl_xor(p1, off, 32);
  }
  if ((tid&31)==0){ r0[tid>>5]=p0; r1[tid>>5]=p1; }
  __syncthreads();
  if (tid==0){
    float a=0.f, c=0.f;
#pragma unroll
    for (int i=0;i<8;++i){ a+=r0[i]; c+=r1[i]; }
    out[b*2+0] = a + b2[0];
    out[b*2+1] = c + b2[1];
  }
}

// ---------------------------------------------------------------- launch
extern "C" void kernel_launch(void* const* d_in, const int* in_sizes, int n_in,
                              void* d_out, int out_size, void* d_ws, size_t ws_size,
                              hipStream_t stream)
{
  (void)in_sizes; (void)n_in; (void)out_size; (void)ws_size;
  const int*   x    = (const int*)d_in[0];
  const unsigned char* mask = (const unsigned char*)d_in[1];
  const float* emb  = (const float*)d_in[2];
  const float* pe   = (const float*)d_in[3];
  const float* Wq   = (const float*)d_in[4];
  const float* bq   = (const float*)d_in[5];
  const float* Wk   = (const float*)d_in[6];
  const float* bk   = (const float*)d_in[7];
  const float* Wv   = (const float*)d_in[8];
  const float* bv   = (const float*)d_in[9];
  const float* Wo   = (const float*)d_in[10];
  const float* bo   = (const float*)d_in[11];
  const float* ln1g = (const float*)d_in[12];
  const float* ln1b = (const float*)d_in[13];
  const float* ln2g = (const float*)d_in[14];
  const float* ln2b = (const float*)d_in[15];
  const float* W1   = (const float*)d_in[16];
  const float* b1   = (const float*)d_in[17];
  const float* W2   = (const float*)d_in[18];
  const float* b2   = (const float*)d_in[19];
  const float* lnfg = (const float*)d_in[20];
  const float* lnfb = (const float*)d_in[21];
  const float* cW1  = (const float*)d_in[22];
  const float* cb1  = (const float*)d_in[23];
  const float* cW2  = (const float*)d_in[24];
  const float* cb2  = (const float*)d_in[25];
  float* out = (float*)d_out;

  const size_t M = TC_M, D = TC_D, DFF = TC_DFF;
  char* ws = (char*)d_ws;
  size_t off = 0;
  auto alloc = [&](size_t bytes)->char*{
    char* p = ws + off; off += (bytes + 255) & ~(size_t)255; return p;
  };
  float* hf32 = (float*)alloc(M*D*4);       // residual stream
  h16*  xf16  = (h16*)alloc(M*D*2);         // LN output (GEMM A)
  h16*  qf16  = (h16*)alloc(M*D*2);
  h16*  kf16  = (h16*)alloc(M*D*2);
  h16*  vf16  = (h16*)alloc(M*D*2);
  h16*  cf16  = (h16*)alloc(M*D*2);         // attention context
  h16*  ff16  = (h16*)alloc(M*DFF*2);       // GELU(FF1); reused as f32 LN out
  h16*  wq16  = (h16*)alloc((size_t)TC_NL*D*D*2);
  h16*  wk16  = (h16*)alloc((size_t)TC_NL*D*D*2);
  h16*  wv16  = (h16*)alloc((size_t)TC_NL*D*D*2);
  h16*  wo16  = (h16*)alloc((size_t)TC_NL*D*D*2);
  h16*  w116  = (h16*)alloc((size_t)TC_NL*D*DFF*2);
  h16*  w216  = (h16*)alloc((size_t)TC_NL*DFF*D*2);
  float* pooled = (float*)alloc((size_t)TC_B*D*4);
  float* hn32 = (float*)ff16;               // final-LN f32 (ff16 dead by then)

  // token embedding + positional encoding
  embed_kernel<<<TC_B*TC_L, 128, 0, stream>>>(x, emb, pe, hf32);

  // weight conversion to f16 (deterministic each call; graph-capture safe)
  const int nDD  = TC_NL*(int)(D*D);
  const int nDF  = TC_NL*(int)(D*DFF);
  f32_to_f16_kernel<<<2048,256,0,stream>>>(Wq, wq16, nDD);
  f32_to_f16_kernel<<<2048,256,0,stream>>>(Wk, wk16, nDD);
  f32_to_f16_kernel<<<2048,256,0,stream>>>(Wv, wv16, nDD);
  f32_to_f16_kernel<<<2048,256,0,stream>>>(Wo, wo16, nDD);
  f32_to_f16_kernel<<<4096,256,0,stream>>>(W1, w116, nDF);
  f32_to_f16_kernel<<<4096,256,0,stream>>>(W2, w216, nDF);

  const dim3 g512((unsigned)(M/128), (unsigned)(D/64));
  const dim3 gff ((unsigned)(M/128), (unsigned)(DFF/64));
  const dim3 gatt((unsigned)(TC_B*TC_H), (unsigned)(TC_L/128));

  for (int i = 0; i < TC_NL; ++i){
    const size_t oDD = (size_t)i*D*D, oDF = (size_t)i*D*DFF, oD = (size_t)i*D;
    layernorm_kernel<h16><<<(unsigned)M,256,0,stream>>>(hf32, ln1g+oD, ln1b+oD, xf16);
    gemm_f16_wmma<EPI_F16><<<g512,256,0,stream>>>(xf16, wq16+oDD, bq+oD, nullptr, qf16, nullptr, (int)M,(int)D,(int)D);
    gemm_f16_wmma<EPI_F16><<<g512,256,0,stream>>>(xf16, wk16+oDD, bk+oD, nullptr, kf16, nullptr, (int)M,(int)D,(int)D);
    gemm_f16_wmma<EPI_F16><<<g512,256,0,stream>>>(xf16, wv16+oDD, bv+oD, nullptr, vf16, nullptr, (int)M,(int)D,(int)D);
    attention_kernel<<<gatt,256,0,stream>>>(qf16, kf16, vf16, mask, cf16);
    gemm_f16_wmma<EPI_RES><<<g512,256,0,stream>>>(cf16, wo16+oDD, bo+oD, hf32, nullptr, hf32, (int)M,(int)D,(int)D);
    layernorm_kernel<h16><<<(unsigned)M,256,0,stream>>>(hf32, ln2g+oD, ln2b+oD, xf16);
    gemm_f16_wmma<EPI_GELU><<<gff,256,0,stream>>>(xf16, w116+oDF, b1+(size_t)i*TC_DFF, nullptr, ff16, nullptr, (int)M,(int)TC_DFF,(int)D);
    gemm_f16_wmma<EPI_RES><<<g512,256,0,stream>>>(ff16, w216+oDF, b2+oD, hf32, nullptr, hf32, (int)M,(int)D,(int)TC_DFF);
  }

  layernorm_kernel<float><<<(unsigned)M,256,0,stream>>>(hf32, lnfg, lnfb, hn32);
  pool_kernel<<<(TC_B*(int)D)/256,256,0,stream>>>(hn32, pooled);
  classifier_kernel<<<TC_B,256,0,stream>>>(pooled, cW1, cb1, cW2, cb2, out);
}
